// Quantize_89928025243834
// MI455X (gfx1250) — compile-verified, matrix operand order checked by reading, not run on hardware
//
#include <hip/hip_runtime.h>

// Problem sizes (fixed by the reference)
#define D      512
#define NROWS  32768
#define MCODES 8192
#define BN     64                 // x rows per workgroup
#define BM     64                 // codebook rows per iteration
#define NITER  (MCODES / BM)      // 128

#define AS1 __attribute__((address_space(1)))
#define AS3 __attribute__((address_space(3)))

#if defined(__AMDGCN__) && __has_builtin(__builtin_amdgcn_global_load_async_to_lds_b128)
#define HAVE_ASYNC_LDS 1
#else
#define HAVE_ASYNC_LDS 0
#endif

typedef __attribute__((ext_vector_type(16))) __bf16 v16bf;
typedef __attribute__((ext_vector_type(8)))  float  v8f;
typedef int v4i __attribute__((vector_size(16)));   // matches async-LDS builtin param

union BFrag {            // one 16x32 bf16 WMMA operand = 8 VGPRs = 2 x b128 LDS loads
    v16bf v;
    uint4 q[2];
};

__device__ __forceinline__ unsigned short f2bf(float f) {  // fp32 -> bf16 RNE
    unsigned u = __float_as_uint(f);
    u += 0x7FFFu + ((u >> 16) & 1u);
    return (unsigned short)(u >> 16);
}

__device__ __forceinline__ void async_wait0() {
#if __has_builtin(__builtin_amdgcn_s_wait_asynccnt)
    __builtin_amdgcn_s_wait_asynccnt(0);
#else
    asm volatile("s_wait_asynccnt 0" ::: "memory");
#endif
}

// ---------------------------------------------------------------------------
// Prep: per-row sum-of-squares for x and W; bf16 copies of W (row-major and
// transposed) into workspace so the main loop can async-DMA them into LDS;
// zero the loss accumulator.
// ---------------------------------------------------------------------------
__global__ void vq_prep_kernel(const float* __restrict__ x,
                               const float* __restrict__ w,
                               float* __restrict__ x2,
                               float* __restrict__ w2,
                               unsigned short* __restrict__ wbf,   // [MCODES][D]
                               unsigned short* __restrict__ wbfT,  // [D][MCODES]
                               float* __restrict__ lossPtr) {
    int r = blockIdx.x * blockDim.x + threadIdx.x;
    if (r == 0) *lossPtr = 0.0f;
    if (r < NROWS) {
        const float4* p = (const float4*)x + (size_t)r * (D / 4);
        float s = 0.0f;
        for (int i = 0; i < D / 4; ++i) {
            float4 v = p[i];
            s += v.x * v.x + v.y * v.y + v.z * v.z + v.w * v.w;
        }
        x2[r] = s;
    } else if (r < NROWS + MCODES) {
        int row = r - NROWS;
        const float4* p = (const float4*)w + (size_t)row * (D / 4);
        float s = 0.0f;
        for (int i = 0; i < D / 4; ++i) {
            float4 v = p[i];
            s += v.x * v.x + v.y * v.y + v.z * v.z + v.w * v.w;
            unsigned short b0 = f2bf(v.x), b1 = f2bf(v.y), b2 = f2bf(v.z), b3 = f2bf(v.w);
            *(uint2*)&wbf[(size_t)row * D + i * 4] =
                make_uint2((unsigned)b0 | ((unsigned)b1 << 16),
                           (unsigned)b2 | ((unsigned)b3 << 16));
            int d0 = i * 4;   // transposed copy: coalesced across the wave (adjacent rows)
            wbfT[(size_t)(d0 + 0) * MCODES + row] = b0;
            wbfT[(size_t)(d0 + 1) * MCODES + row] = b1;
            wbfT[(size_t)(d0 + 2) * MCODES + row] = b2;
            wbfT[(size_t)(d0 + 3) * MCODES + row] = b3;
        }
        w2[row] = s;
    }
}

// ---------------------------------------------------------------------------
// Fused flash-style soft-VQ kernel.
// 8 waves: wave w owns output rows rt=w&3 (16 rows) x cols ch=w>>2 (256 cols).
// ---------------------------------------------------------------------------
__global__ __launch_bounds__(256, 1)
void vq_fused_kernel(const float* __restrict__ x,
                     const float* __restrict__ w,
                     const int*   __restrict__ tptr,
                     const float* __restrict__ x2g,
                     const float* __restrict__ w2g,
                     const unsigned short* __restrict__ wbf,
                     const unsigned short* __restrict__ wbfT,
                     float* __restrict__ outQ,
                     float* __restrict__ outIds,
                     float* __restrict__ outLoss) {
    __shared__ __align__(16) unsigned short Xs[BN * D];    // 64 KB bf16 x tile [row][d]
    __shared__ __align__(16) unsigned short Ws[BM * D];    // 64 KB bf16 W tile [code][d]
    __shared__ __align__(16) unsigned short Wt[D * BM];    // 64 KB bf16 W tile [d][code]
    __shared__ __align__(16) float          Ss[BN * BM];   // 16 KB scores
    __shared__ __align__(16) unsigned short Ps[BN * BM];   //  8 KB softmax probs (bf16)
    __shared__ float x2s[BN];
    __shared__ float w2s[BM];
    __shared__ float alphas[BN];
    __shared__ float lrow_s[BN];
    __shared__ int   besti_s[BN];
    __shared__ float red[256];

    const int tid  = threadIdx.x;
    const int wv   = tid >> 5;
    const int lane = tid & 31;
    const int rt   = wv & 3;          // row-tile (16 rows) owned by this wave
    const int ch   = wv >> 2;         // column half (256 cols) owned by this wave
    const int l15  = lane & 15;
    const int lh   = lane >> 4;       // K-half within WMMA operand layout
    const int xbase = blockIdx.x * BN;

    const float invt = 1.0f / (float)tptr[0];

    // ---- load x tile once, convert to bf16 in LDS -------------------------
    const float4* xg4 = (const float4*)x;
    for (int p = 0; p < 32; ++p) {
        int idx = p * 256 + tid;               // float4 index into tile
        int row = idx >> 7;                    // 128 float4 per row
        int c4  = idx & 127;
        float4 v = xg4[(size_t)(xbase + row) * (D / 4) + c4];
        unsigned u0 = (unsigned)f2bf(v.x) | ((unsigned)f2bf(v.y) << 16);
        unsigned u1 = (unsigned)f2bf(v.z) | ((unsigned)f2bf(v.w) << 16);
        *(uint2*)&Xs[row * D + c4 * 4] = make_uint2(u0, u1);
    }
    if (tid < BN) x2s[tid] = x2g[xbase + tid];

    // per-wave fp32 accumulator: 16 tiles of 16x16 => 128 VGPRs / lane
    v8f acc[16];
#pragma unroll
    for (int t = 0; t < 16; ++t)
#pragma unroll
        for (int j = 0; j < 8; ++j) acc[t][j] = 0.0f;

    // online-softmax per-row state (threads 0..63 own row tid)
    float m_r = -3.0e38f, l_r = 0.0f;
    int   b_r = 0;

    for (int it = 0; it < NITER; ++it) {
        const int mbase = it * BM;

        // ---- W tile -> LDS -----------------------------------------------
#if HAVE_ASYNC_LDS
        {
            // row-major tile: contiguous 64 KB starting at wbf[mbase*D]
            const unsigned short* wsrc = wbf + (size_t)mbase * D;
#pragma unroll
            for (int c = 0; c < 16; ++c) {
                int off = (c * 256 + tid) * 8;          // ushort offset, 16B chunks
                __builtin_amdgcn_global_load_async_to_lds_b128(
                    (AS1 v4i*)(wsrc + off), (AS3 v4i*)&Ws[off], 0, 0);
            }
            // transposed tile: 512 rows of 64 codes (128B each) from wbfT
#pragma unroll
            for (int c = 0; c < 16; ++c) {
                int idx = c * 256 + tid;                // 16B chunk id
                int dd  = idx >> 3;                     // 8 chunks per d-row
                int s8  = (idx & 7) * 8;                // ushort offset within row
                __builtin_amdgcn_global_load_async_to_lds_b128(
                    (AS1 v4i*)(wbfT + (size_t)dd * MCODES + mbase + s8),
                    (AS3 v4i*)&Wt[dd * BM + s8], 0, 0);
            }
            if (it + 1 < NITER)  // warm L2 for the next tile
                __builtin_prefetch(wbf + (size_t)(mbase + BM) * D + tid * 8, 0, 0);
            async_wait0();
        }
#else
        {
            const float4* wg4 = (const float4*)w;
            for (int p = 0; p < 32; ++p) {
                int idx = p * 256 + tid;
                int row = idx >> 7;
                int c4  = idx & 127;
                float4 v = wg4[(size_t)(mbase + row) * (D / 4) + c4];
                unsigned short b0 = f2bf(v.x), b1 = f2bf(v.y), b2 = f2bf(v.z), b3 = f2bf(v.w);
                *(uint2*)&Ws[row * D + c4 * 4] =
                    make_uint2((unsigned)b0 | ((unsigned)b1 << 16),
                               (unsigned)b2 | ((unsigned)b3 << 16));
                int d0 = c4 * 4;
                Wt[(d0 + 0) * BM + row] = b0;
                Wt[(d0 + 1) * BM + row] = b1;
                Wt[(d0 + 2) * BM + row] = b2;
                Wt[(d0 + 3) * BM + row] = b3;
            }
        }
#endif
        if (tid < BM) w2s[tid] = w2g[mbase + tid];
        __syncthreads();

        // ---- Phase A: S[64,64] = x . W^T  (bf16 WMMA, K = 512) -----------
        {
            const int ct0 = ch * 2;   // this wave computes col-tiles ct0, ct0+1
            v8f s0, s1;
#pragma unroll
            for (int j = 0; j < 8; ++j) { s0[j] = 0.0f; s1[j] = 0.0f; }
            const unsigned short* arow  = &Xs[(rt * 16 + l15) * D + lh * 8];
            const unsigned short* b0row = &Ws[(ct0 * 16 + l15) * D + lh * 8];
            const unsigned short* b1row = &Ws[((ct0 + 1) * 16 + l15) * D + lh * 8];
#pragma unroll 4
            for (int k = 0; k < D; k += 32) {
                BFrag a, b0, b1;
                a.q[0]  = *(const uint4*)(arow + k);
                a.q[1]  = *(const uint4*)(arow + k + 16);
                b0.q[0] = *(const uint4*)(b0row + k);
                b0.q[1] = *(const uint4*)(b0row + k + 16);
                b1.q[0] = *(const uint4*)(b1row + k);
                b1.q[1] = *(const uint4*)(b1row + k + 16);
                s0 = __builtin_amdgcn_wmma_f32_16x16x32_bf16(false, a.v, false, b0.v,
                                                             (short)0, s0, false, false);
                s1 = __builtin_amdgcn_wmma_f32_16x16x32_bf16(false, a.v, false, b1.v,
                                                             (short)0, s1, false, false);
            }
            // scores = -dist/t, written to LDS
#pragma unroll
            for (int j = 0; j < 8; ++j) {
                int rl  = rt * 16 + lh * 8 + j;
                int cl0 = ct0 * 16 + l15;
                int cl1 = cl0 + 16;
                float d2a = x2s[rl] + w2s[cl0] - 2.0f * s0[j];
                float d2b = x2s[rl] + w2s[cl1] - 2.0f * s1[j];
                Ss[rl * BM + cl0] = -sqrtf(fmaxf(d2a, 0.0f)) * invt;
                Ss[rl * BM + cl1] = -sqrtf(fmaxf(d2b, 0.0f)) * invt;
            }
        }
        __syncthreads();

        // ---- Phase B: online softmax, one thread per row -----------------
        if (tid < BN) {
            const float* srow = &Ss[tid * BM];
            float tm = -3.0e38f; int ta = 0;
            for (int j = 0; j < BM; ++j) {
                float v = srow[j];
                if (v > tm) { tm = v; ta = j; }
            }
            float mold = m_r;
            float mnew = fmaxf(mold, tm);
            float alpha = __expf(mold - mnew);
            float sum = 0.0f;
            for (int j = 0; j < BM; ++j) {
                float pv = __expf(srow[j] - mnew);
                sum += pv;
                Ps[tid * BM + j] = f2bf(pv);
            }
            l_r = l_r * alpha + sum;
            if (tm > mold) b_r = mbase + ta;   // argmax score == argmin distance
            m_r = mnew;
            alphas[tid] = alpha;
        }
        __syncthreads();

        // ---- Phase C: O = O*alpha + P . W  (K = 64) ----------------------
        {
            float av[8];
#pragma unroll
            for (int j = 0; j < 8; ++j) av[j] = alphas[rt * 16 + lh * 8 + j];
            BFrag pa0, pa1;
            const unsigned short* prow = &Ps[(rt * 16 + l15) * BM + lh * 8];
            pa0.q[0] = *(const uint4*)(prow);
            pa0.q[1] = *(const uint4*)(prow + 16);
            pa1.q[0] = *(const uint4*)(prow + 32);
            pa1.q[1] = *(const uint4*)(prow + 48);
#pragma unroll
            for (int t = 0; t < 16; ++t) {
                int dcol = ch * 256 + t * 16 + l15;
                const unsigned short* brow = &Wt[dcol * BM + lh * 8];
                BFrag b0, b1;
                b0.q[0] = *(const uint4*)(brow);
                b0.q[1] = *(const uint4*)(brow + 16);
                b1.q[0] = *(const uint4*)(brow + 32);
                b1.q[1] = *(const uint4*)(brow + 48);
                v8f c = acc[t];
#pragma unroll
                for (int j = 0; j < 8; ++j) c[j] *= av[j];
                c = __builtin_amdgcn_wmma_f32_16x16x32_bf16(false, pa0.v, false, b0.v,
                                                            (short)0, c, false, false);
                c = __builtin_amdgcn_wmma_f32_16x16x32_bf16(false, pa1.v, false, b1.v,
                                                            (short)0, c, false, false);
                acc[t] = c;
            }
        }
        __syncthreads();
    }

    // ---- epilogue: normalize, STE output, fused 1.25*MSE, ids ------------
    if (tid < BN) { lrow_s[tid] = l_r; besti_s[tid] = b_r; }
    __syncthreads();

    float lv[8];
#pragma unroll
    for (int j = 0; j < 8; ++j) lv[j] = lrow_s[rt * 16 + lh * 8 + j];

    float lloss = 0.0f;
#pragma unroll
    for (int t = 0; t < 16; ++t) {
        int gcol = ch * 256 + t * 16 + l15;
#pragma unroll
        for (int j = 0; j < 8; ++j) {
            int grow = xbase + rt * 16 + lh * 8 + j;
            float q  = acc[t][j] / lv[j];
            float xv = x[(size_t)grow * D + gcol];
            float dq = q - xv;
            lloss += dq * dq;
            outQ[(size_t)grow * D + gcol] = xv + (q - xv);   // straight-through value
        }
    }
    red[tid] = lloss;
    __syncthreads();
    for (int s = 128; s > 0; s >>= 1) {
        if (tid < s) red[tid] += red[tid + s];
        __syncthreads();
    }
    if (tid == 0)
        atomicAdd(outLoss, red[0] * (1.25f / ((float)NROWS * (float)D)));
    if (tid < BN) outIds[xbase + tid] = (float)besti_s[tid];
}

// ---------------------------------------------------------------------------
extern "C" void kernel_launch(void* const* d_in, const int* in_sizes, int n_in,
                              void* d_out, int out_size, void* d_ws, size_t ws_size,
                              hipStream_t stream) {
    const float* x = (const float*)d_in[0];
    const float* w = (const float*)d_in[1];
    const int*   t = (const int*)d_in[2];

    float* ws_x2 = (float*)d_ws;                               // [NROWS]
    float* ws_w2 = ws_x2 + NROWS;                              // [MCODES]
    unsigned short* ws_wbf  = (unsigned short*)(ws_w2 + MCODES);   // [MCODES*D] bf16
    unsigned short* ws_wbfT = ws_wbf + (size_t)MCODES * D;         // [D*MCODES] bf16

    float* outQ    = (float*)d_out;                      // [NROWS*D]
    float* outIds  = outQ + (size_t)NROWS * D;           // [NROWS]
    float* outLoss = outIds + NROWS;                     // [1]

    int rows = NROWS + MCODES;
    vq_prep_kernel<<<(rows + 255) / 256, 256, 0, stream>>>(x, w, ws_x2, ws_w2,
                                                           ws_wbf, ws_wbfT, outLoss);
    vq_fused_kernel<<<NROWS / BN, 256, 0, stream>>>(x, w, t, ws_x2, ws_w2,
                                                    ws_wbf, ws_wbfT,
                                                    outQ, outIds, outLoss);
}
